// Graph_ECC_71811853189357
// MI455X (gfx1250) — matrix operand order repaired and checked by reading, not compiled, not run on hardware
//
#include <hip/hip_runtime.h>
#include <hip/hip_bf16.h>

#define NNODES 8192
#define NEDGES 32768

typedef __attribute__((ext_vector_type(16))) __bf16 v16bf;
typedef __attribute__((ext_vector_type(8)))  float  v8f;
typedef __attribute__((ext_vector_type(4)))  int    v4i;

#if defined(__HIP_DEVICE_COMPILE__) && defined(__has_builtin)
#  if __has_builtin(__builtin_amdgcn_global_load_async_to_lds_b128) && \
      __has_builtin(__builtin_amdgcn_s_wait_asynccnt)
#    define HAVE_ASYNC_LDS 1
#  endif
#endif
#ifndef HAVE_ASYNC_LDS
#  define HAVE_ASYNC_LDS 0
#endif

union BReg { v16bf v; uint4 q[2]; unsigned short h[16]; };

__device__ inline unsigned short f2bf(float f) {
  union { float f; unsigned u; } cv; cv.f = f;
  unsigned r = cv.u + 0x7fffu + ((cv.u >> 16) & 1u);
  return (unsigned short)(r >> 16);
}
__device__ inline uint4 u4z() { uint4 z; z.x = z.y = z.z = z.w = 0u; return z; }
__device__ inline float lrelu(float v) { return v > 0.f ? v : 0.01f * v; }
__device__ inline unsigned hashu(unsigned x) {
  x ^= x >> 16; x *= 0x7feb352dU; x ^= x >> 15; x *= 0x846ca68bU; x ^= x >> 16; return x;
}

// Stage one 16B chunk global -> LDS (async DMA if available, else via VGPRs).
__device__ inline void stage_chunk(const uint4* g, uint4* l) {
#if HAVE_ASYNC_LDS
  __builtin_amdgcn_global_load_async_to_lds_b128(
      (v4i*)(void*)g,
      (__attribute__((address_space(3))) v4i*)l, 0, 0);
#else
  *l = *g;
#endif
}
__device__ inline void stage_wait() {
#if HAVE_ASYNC_LDS
  __builtin_amdgcn_s_wait_asynccnt(0);
#endif
}

__global__ void k_zero(float* p, int n) {
  int i = blockIdx.x * blockDim.x + threadIdx.x;
  if (i < n) p[i] = 0.f;
}

__global__ void k_count(const int* __restrict__ ei, float* __restrict__ cnt, int ne) {
  int e = blockIdx.x * blockDim.x + threadIdx.x;
  if (e < ne) atomicAdd(cnt + ei[ne + e], 1.0f);
}

// Repack wb [hid, ncols] f32 -> WMMA-B lane order, bf16, K padded to 64 with zeros.
// For tile t (16 cols), K-step s (32 K): lane holds col = t*16+(lane&15),
// K = s*32 + (lane>>4)*16 + j (j=0..15); 16 bf16 = 32 contiguous bytes per lane.
__global__ void k_pack_wb(const float* __restrict__ wb, unsigned short* __restrict__ bp,
                          int ncols, int hid) {
  int id = blockIdx.x * blockDim.x + threadIdx.x;
  if (id >= ncols * 64) return;
  int j = id & 15, lane = (id >> 4) & 31, s = (id >> 9) & 1, t = id >> 10;
  int col = t * 16 + (lane & 15);
  int k = s * 32 + ((lane >> 4) << 4) + j;
  float v = (k < hid) ? wb[k * ncols + col] : 0.f;
  bp[id] = f2bf(v);
}

// Edge-MLP first layer: a = lrelu(ea @ wa + ba), stored row-major bf16 [E,64] (cols>=hid zero).
// One wave per 16-edge x 16-col tile; K=16 padded to 32 (hi lanes carry zero K-half).
__global__ __launch_bounds__(32)
void k_edge_hidden(const float* __restrict__ ea, const float* __restrict__ wa,
                   const float* __restrict__ ba, unsigned short* __restrict__ abuf, int hid) {
  int tile = blockIdx.x, nt = blockIdx.y;
  int lane = threadIdx.x, m = lane & 15;
  int hi = (lane >= 16) ? 1 : 0;
  int n0 = nt * 16;
  int e = tile * 16 + m;
  if (n0 >= hid) {                       // zero-pad columns beyond hid
    if (!hi) {
      uint4 z = u4z();
      *(uint4*)(abuf + (size_t)e * 64 + n0)     = z;
      *(uint4*)(abuf + (size_t)e * 64 + n0 + 8) = z;
    }
    return;
  }
  BReg A, B;
  if (!hi) {
    const float* ar = ea + (size_t)e * 16;
#pragma unroll
    for (int k = 0; k < 16; ++k) A.h[k] = f2bf(ar[k]);           // K=0..15, row e
    int n = n0 + m;
#pragma unroll
    for (int k = 0; k < 16; ++k) B.h[k] = f2bf(wa[k * hid + n]); // K=0..15, col n
  } else {
    A.q[0] = u4z(); A.q[1] = u4z();       // K=16..31 is zero padding
    B.q[0] = u4z(); B.q[1] = u4z();
  }
  v8f c = {};
  c = __builtin_amdgcn_wmma_f32_16x16x32_bf16(false, A.v, false, B.v, (short)0, c, false, false);
  int col = n0 + m;
#pragma unroll
  for (int r = 0; r < 8; ++r) {
    int row = r + hi * 8;
    float v = lrelu(c[r] + ba[col]);
    abuf[(size_t)(tile * 16 + row) * 64 + col] = f2bf(v);
  }
}

// Fused message kernel, 8 waves / 128 edges per block. Per input channel i the 8 waves
// consume the SAME B-slab (OT*2 KiB of packed wb), so it is double-buffered into LDS
// (async-to-LDS DMA when available) and read back with ds_load_b128; each wave's W tile
// is produced by 2x v_wmma_f32_16x16x32_bf16 (K=64) and consumed in-register:
//   msg[e,o] += h[src[e],i] * (W[e,i,o] + bb[i,o]);  then scatter-add (atomics, L2-resident).
template<int IN_C, int OUT_C>
__global__ __launch_bounds__(256)
void k_edge_msg(const unsigned short* __restrict__ abuf,
                const unsigned short* __restrict__ bpack,
                const float* __restrict__ bb,
                const float* __restrict__ hsrc,
                const int* __restrict__ ei,
                float* __restrict__ osum) {
  constexpr int OT   = OUT_C / 16;
  constexpr int SLAB = OT * 128;          // uint4 chunks per i-slab (OT*2048 bytes)
  int tid = threadIdx.x;
  int w = tid >> 5, lane = tid & 31;
  int m = lane & 15;
  int hi = (lane >= 16) ? 1 : 0;
  int ebase = blockIdx.x * 128;
  int e = ebase + w * 16 + m;

  __shared__ float sh_h[128 * IN_C];
  __shared__ int   sh_dst[128];
  __shared__ uint4 shB[2][SLAB];

  // gather h[src] rows into LDS: 2 threads per edge row
  {
    int row = tid >> 1;
    int er = ebase + row;
    int s = ei[er];
    const float* hr = hsrc + (size_t)s * IN_C;
    for (int i = tid & 1; i < IN_C; i += 2) sh_h[row * IN_C + i] = hr[i];
    if ((tid & 1) == 0) sh_dst[row] = ei[NEDGES + er];
  }

  // A operand (edge activations), 2 K-steps; contiguous 16-bit A layout:
  // lane<16: K s*32+{0..15}; lane>=16: +16.
  const uint4* aq = (const uint4*)(abuf + (size_t)e * 64);
  BReg A0, A1;
  A0.q[0] = aq[0 + 2 * hi]; A0.q[1] = aq[1 + 2 * hi];
  A1.q[0] = aq[4 + 2 * hi]; A1.q[1] = aq[5 + 2 * hi];

  const uint4* bq = (const uint4*)bpack;

  // prologue: stage slab 0 into buffer 0
  for (int c = tid; c < SLAB; c += 256) stage_chunk(bq + (size_t)0 * SLAB + c, &shB[0][c]);
  stage_wait();
  __syncthreads();   // slab 0 + sh_h ready

  float acc[OT][8];
#pragma unroll
  for (int ot = 0; ot < OT; ++ot)
#pragma unroll
    for (int r = 0; r < 8; ++r) acc[ot][r] = 0.f;

  for (int i = 0; i < IN_C; ++i) {
    int cur = i & 1;
    if (i + 1 < IN_C)  // prefetch next slab into the other buffer
      for (int c = tid; c < SLAB; c += 256)
        stage_chunk(bq + (size_t)(i + 1) * SLAB + c, &shB[cur ^ 1][c]);

    float hv[8];
#pragma unroll
    for (int r = 0; r < 8; ++r)
      hv[r] = sh_h[(w * 16 + r + hi * 8) * IN_C + i];   // LDS broadcast

#pragma unroll
    for (int ot = 0; ot < OT; ++ot) {
      const uint4* bp = &shB[cur][(ot * 2 + 0) * 64 + lane * 2];  // ds_load_b128 x4
      BReg B0, B1;
      B0.q[0] = bp[0];  B0.q[1] = bp[1];
      B1.q[0] = bp[64]; B1.q[1] = bp[65];
      v8f c = {};
      c = __builtin_amdgcn_wmma_f32_16x16x32_bf16(false, A0.v, false, B0.v, (short)0, c, false, false);
      c = __builtin_amdgcn_wmma_f32_16x16x32_bf16(false, A1.v, false, B1.v, (short)0, c, false, false);
      float bbv = bb[i * OUT_C + ot * 16 + m];
#pragma unroll
      for (int r = 0; r < 8; ++r) acc[ot][r] += hv[r] * (c[r] + bbv);
    }

    stage_wait();      // own async fills of next buffer complete
    __syncthreads();   // all waves done reading shB[cur] and done staging
  }

#pragma unroll
  for (int ot = 0; ot < OT; ++ot)
#pragma unroll
    for (int r = 0; r < 8; ++r) {
      int d = sh_dst[w * 16 + r + hi * 8];
      atomicAdd(osum + (size_t)d * OUT_C + ot * 16 + m, acc[ot][r]);
    }
}

// Node combine: mean + h@root + bias, LeakyReLU, optional concat with x.
template<int IN_C, int OUT_C, int CONCAT>
__global__ __launch_bounds__(128)
void k_node(const float* __restrict__ sum, const float* __restrict__ cnt,
            const float* __restrict__ h, const float* __restrict__ root,
            const float* __restrict__ bias, const float* __restrict__ x,
            float* __restrict__ out) {
  int n = blockIdx.x, t = threadIdx.x;
  __shared__ float sh[IN_C];
  if (t < IN_C) sh[t] = h[(size_t)n * IN_C + t];
  __syncthreads();
  constexpr int OW = OUT_C + (CONCAT ? 16 : 0);
  if (t < OUT_C) {
    float c = cnt[n]; c = c < 1.f ? 1.f : c;
    float v = sum[(size_t)n * OUT_C + t] / c + bias[t];
    for (int i = 0; i < IN_C; ++i) v += sh[i] * root[i * OUT_C + t];
    out[(size_t)n * OW + t] = lrelu(v);
  } else if (CONCAT && t < OUT_C + 16) {
    out[(size_t)n * OW + t] = x[n * 16 + (t - OUT_C)];
  }
}

// Hard gumbel-softmax with tau=-1: logits = -(d3+g); output = one-hot(argmax(soft)).
__global__ void k_gumbel(const float* __restrict__ d3, float* __restrict__ out, int nn) {
  int n = blockIdx.x * blockDim.x + threadIdx.x;
  if (n >= nn) return;
  float l[16], mx = -1e30f;
#pragma unroll
  for (int j = 0; j < 16; ++j) {
    unsigned hsh = hashu((unsigned)(n * 16 + j) * 2654435761u + 42u);
    float u = (float)(hsh >> 8) * (1.0f / 16777216.0f) + 1e-12f;
    float g = -__logf(-__logf(u));
    l[j] = -(d3[n * 16 + j] + g);
    mx = fmaxf(mx, l[j]);
  }
  float s = 0.f, ex[16];
#pragma unroll
  for (int j = 0; j < 16; ++j) { ex[j] = __expf(l[j] - mx); s += ex[j]; }
  int arg = 0; float best = -1.f;
#pragma unroll
  for (int j = 0; j < 16; ++j) { float sf = ex[j] / s; if (sf > best) { best = sf; arg = j; } }
#pragma unroll
  for (int j = 0; j < 16; ++j) {
    float sf = ex[j] / s;
    out[n * 16 + j] = (j == arg ? 1.f : 0.f) + sf - sf;  // == hard
  }
}

extern "C" void kernel_launch(void* const* d_in, const int* in_sizes, int n_in,
                              void* d_out, int out_size, void* d_ws, size_t ws_size,
                              hipStream_t stream) {
  (void)in_sizes; (void)n_in; (void)out_size; (void)ws_size;
  const float* x    = (const float*)d_in[0];
  const int*   ei   = (const int*)d_in[1];
  const float* ea   = (const float*)d_in[2];
  const float* w1a  = (const float*)d_in[3];
  const float* b1a  = (const float*)d_in[4];
  const float* w1b  = (const float*)d_in[5];
  const float* b1b  = (const float*)d_in[6];
  const float* root1= (const float*)d_in[7];
  const float* bias1= (const float*)d_in[8];
  const float* w2a  = (const float*)d_in[9];
  const float* b2a  = (const float*)d_in[10];
  const float* w2b  = (const float*)d_in[11];
  const float* b2b  = (const float*)d_in[12];
  const float* root2= (const float*)d_in[13];
  const float* bias2= (const float*)d_in[14];
  const float* w3a  = (const float*)d_in[15];
  const float* b3a  = (const float*)d_in[16];
  const float* w3b  = (const float*)d_in[17];
  const float* b3b  = (const float*)d_in[18];
  const float* root3= (const float*)d_in[19];
  const float* bias3= (const float*)d_in[20];

  char* ws = (char*)d_ws;
  unsigned short* ABUF = (unsigned short*)(ws + 0);            // E*64 bf16 = 4 MiB
  unsigned short* WBP  = (unsigned short*)(ws + 4194304);      // <=64*5120 bf16 (768 KiB slot)
  float* H2  = (float*)(ws + 4980736);                         // N*80 f32
  float* H3  = (float*)(ws + 7602176);                         // N*80 f32
  float* D3  = (float*)(ws + 10223616);                        // N*16 f32
  float* SUM = (float*)(ws + 10747904);                        // N*64 f32
  float* CNT = (float*)(ws + 12845056);                        // N f32

  // in-degree counts (same for all layers)
  k_zero<<<(NNODES + 255) / 256, 256, 0, stream>>>(CNT, NNODES);
  k_count<<<(NEDGES + 255) / 256, 256, 0, stream>>>(ei, CNT, NEDGES);

  // ---- Layer 1: in=16, out=64, hid=32 ----
  k_pack_wb<<<(1024 * 64 + 255) / 256, 256, 0, stream>>>(w1b, WBP, 1024, 32);
  k_edge_hidden<<<dim3(NEDGES / 16, 4), 32, 0, stream>>>(ea, w1a, b1a, ABUF, 32);
  k_zero<<<(NNODES * 64 + 255) / 256, 256, 0, stream>>>(SUM, NNODES * 64);
  k_edge_msg<16, 64><<<NEDGES / 128, 256, 0, stream>>>(ABUF, WBP, b1b, x, ei, SUM);
  k_node<16, 64, 1><<<NNODES, 128, 0, stream>>>(SUM, CNT, x, root1, bias1, x, H2);

  // ---- Layer 2: in=80, out=64, hid=48 ----
  k_pack_wb<<<(5120 * 64 + 255) / 256, 256, 0, stream>>>(w2b, WBP, 5120, 48);
  k_edge_hidden<<<dim3(NEDGES / 16, 4), 32, 0, stream>>>(ea, w2a, b2a, ABUF, 48);
  k_zero<<<(NNODES * 64 + 255) / 256, 256, 0, stream>>>(SUM, NNODES * 64);
  k_edge_msg<80, 64><<<NEDGES / 128, 256, 0, stream>>>(ABUF, WBP, b2b, H2, ei, SUM);
  k_node<80, 64, 1><<<NNODES, 128, 0, stream>>>(SUM, CNT, H2, root2, bias2, x, H3);

  // ---- Layer 3: in=80, out=16, hid=64 ----
  k_pack_wb<<<(1280 * 64 + 255) / 256, 256, 0, stream>>>(w3b, WBP, 1280, 64);
  k_edge_hidden<<<dim3(NEDGES / 16, 4), 32, 0, stream>>>(ea, w3a, b3a, ABUF, 64);
  k_zero<<<(NNODES * 16 + 255) / 256, 256, 0, stream>>>(SUM, NNODES * 16);
  k_edge_msg<80, 16><<<NEDGES / 128, 256, 0, stream>>>(ABUF, WBP, b3b, H3, ei, SUM);
  k_node<80, 16, 0><<<NNODES, 128, 0, stream>>>(SUM, CNT, H3, root3, bias3, x, D3);

  // ---- Gumbel hard one-hot ----
  k_gumbel<<<(NNODES + 255) / 256, 256, 0, stream>>>(D3, (float*)d_out, NNODES);
}
